// MoELayer_39651138076718
// MI455X (gfx1250) — matrix operand order repaired and checked by reading, not compiled, not run on hardware
//
#include <hip/hip_runtime.h>

typedef float  v8f   __attribute__((ext_vector_type(8)));
typedef __bf16 v4bf  __attribute__((ext_vector_type(4)));
typedef __bf16 v8bf  __attribute__((ext_vector_type(8)));
typedef __bf16 v16bf __attribute__((ext_vector_type(16)));

#define TOKS 128     // tokens per workgroup tile
#define DDIM 128
#define CDIM 128
#define NEXP 8
#define XSTR 132     // f32 x plane stride (floats)
#define HSTR 136     // bf16 hi/lo plane stride (bf16 elems) -> conflict-free b128

// ---- pre-kernel: expert_w f32 [E,D,C] -> transposed bf16 hi/lo [E,C,D] in ws
__global__ __launch_bounds__(256) void convert_w_kernel(
    const float* __restrict__ ew, __bf16* __restrict__ whi, __bf16* __restrict__ wlo)
{
  int idx = blockIdx.x * 256 + threadIdx.x;        // e*16384 + d*128 + c
  int c = idx & 127, d = (idx >> 7) & 127, e = idx >> 14;
  float w = ew[idx];                                // coalesced read
  __bf16 hh = (__bf16)w;
  __bf16 ll = (__bf16)(w - (float)hh);
  int t = (e * CDIM + c) * DDIM + d;                // transposed scatter (L2 OK)
  whi[t] = hh;
  wlo[t] = ll;
}

__global__ __launch_bounds__(256) void moe_fused_kernel(
    const float*  __restrict__ x,          // [B*N, D]
    const float*  __restrict__ router_w,   // [D, E]
    const float*  __restrict__ router_b,   // [E]
    const float*  __restrict__ expert_b,   // [E, C]
    const __bf16* __restrict__ whi,        // [E, C, D] hi plane
    const __bf16* __restrict__ wlo,        // [E, C, D] lo plane
    float* __restrict__ out)               // [B*N, C]
{
  __shared__ float  xs[TOKS * XSTR];       // 67584 B  f32 x (router precision)
  __shared__ __bf16 xhi[TOKS * HSTR];      // 34816 B  bf16 hi plane
  __shared__ __bf16 xlo[TOKS * HSTR];      // 34816 B  bf16 lo plane
  __shared__ float  gates[TOKS * NEXP];    //  4096 B  [token][e]
  __shared__ float  gatesT[NEXP * TOKS];   //  4096 B  [e][token]
  __shared__ float  rwl[DDIM * NEXP];      //  4096 B

  const int tid  = threadIdx.x;
  const int lane = tid & 31;
  const int wave = tid >> 5;               // 8 waves
  const int h    = lane >> 4;              // lane half
  const int l    = lane & 15;
  const long token0 = (long)blockIdx.x * TOKS;

  // ---------------- Phase A: stage x (f32 + bf16 hi/lo) + router_w ----------
  {
    const float4* xg = (const float4*)(x + token0 * DDIM);
    #pragma unroll
    for (int i = 0; i < 16; ++i) {
      int ch  = tid + 256 * i;
      int row = ch >> 5;
      int c4  = ch & 31;
      float4 v = xg[row * 32 + c4];                    // coalesced 16B
      *(float4*)&xs[row * XSTR + c4 * 4] = v;
      v4bf hv, lv;
      float vv[4] = {v.x, v.y, v.z, v.w};
      #pragma unroll
      for (int j = 0; j < 4; ++j) {
        __bf16 hb = (__bf16)vv[j];
        hv[j] = hb;
        lv[j] = (__bf16)(vv[j] - (float)hb);
      }
      *(v4bf*)&xhi[row * HSTR + c4 * 4] = hv;          // 8B aligned
      *(v4bf*)&xlo[row * HSTR + c4 * 4] = lv;
    }
    #pragma unroll
    for (int i = 0; i < 4; ++i)
      rwl[tid + 256 * i] = router_w[tid + 256 * i];
  }
  __syncthreads();

  // ---------------- Phase B: router fp32 (softmax+top2 == sigmoid) ----------
  if (tid < TOKS) {
    float logit[NEXP];
    #pragma unroll
    for (int e = 0; e < NEXP; ++e) logit[e] = router_b[e];
    #pragma unroll 4
    for (int d = 0; d < DDIM; ++d) {
      float xv = xs[tid * XSTR + d];
      #pragma unroll
      for (int e = 0; e < NEXP; ++e)
        logit[e] = fmaf(xv, rwl[d * NEXP + e], logit[e]);
    }
    int i0 = 0;
    #pragma unroll
    for (int e = 1; e < NEXP; ++e) if (logit[e] > logit[i0]) i0 = e;
    int i1 = (i0 == 0) ? 1 : 0;
    #pragma unroll
    for (int e = 0; e < NEXP; ++e)
      if (e != i0 && logit[e] > logit[i1]) i1 = e;
    float g0 = 1.0f / (1.0f + __expf(logit[i1] - logit[i0]));
    float g1 = 1.0f - g0;
    #pragma unroll
    for (int e = 0; e < NEXP; ++e) { gates[tid * NEXP + e] = 0.0f; gatesT[e * TOKS + tid] = 0.0f; }
    gates[tid * NEXP + i0] = g0;  gatesT[i0 * TOKS + tid] = g0;
    gates[tid * NEXP + i1] = g1;  gatesT[i1 * TOKS + tid] = g1;
  }
  __syncthreads();

  // ---------------- Phase C: split-bf16 3-pass WMMA expert GEMMs ------------
  // wave tile: 64 rows (rh) x 32 cols (cg): acc[4 row-tiles][2 col-tiles]
  const int rh = wave >> 2;
  const int cg = wave & 3;
  const int n0 = cg * 32;

  v8f acc[4][2];
  {  // acc init = sum_e gate[row,e] * b[e,col]   (bias of the combine)
    float bc[2][NEXP];
    #pragma unroll
    for (int nt = 0; nt < 2; ++nt)
      #pragma unroll
      for (int e = 0; e < NEXP; ++e)
        bc[nt][e] = expert_b[e * CDIM + n0 + nt * 16 + l];
    #pragma unroll
    for (int mtl = 0; mtl < 4; ++mtl) {
      #pragma unroll
      for (int i = 0; i < 8; ++i) {
        const int row = rh * 64 + mtl * 16 + i + 8 * h;
        const float4 glo = *(const float4*)&gates[row * NEXP];
        const float4 ghi = *(const float4*)&gates[row * NEXP + 4];
        #pragma unroll
        for (int nt = 0; nt < 2; ++nt)
          acc[mtl][nt][i] =
              glo.x * bc[nt][0] + glo.y * bc[nt][1] + glo.z * bc[nt][2] +
              glo.w * bc[nt][3] + ghi.x * bc[nt][4] + ghi.y * bc[nt][5] +
              ghi.z * bc[nt][6] + ghi.w * bc[nt][7];
      }
    }
  }

  #pragma unroll 1
  for (int e = 0; e < NEXP; ++e) {
    const __bf16* Whe = whi + (long)e * DDIM * CDIM;
    const __bf16* Wle = wlo + (long)e * DDIM * CDIM;
    #pragma unroll 1
    for (int kbc = 0; kbc < 4; ++kbc) {    // K blocks of 32
      const int kO = kbc * 32 + 8 * h;     // B frag K: {8h..8h+7, 8h+16..8h+23}
      v16bf Bh[2], Bl[2];
      #pragma unroll
      for (int nt = 0; nt < 2; ++nt) {
        const __bf16* ph = Whe + (long)(n0 + nt * 16 + l) * DDIM + kO;
        const __bf16* pl = Wle + (long)(n0 + nt * 16 + l) * DDIM + kO;
        v8bf b0 = *(const v8bf*)ph, b1 = *(const v8bf*)(ph + 16);
        v8bf c0 = *(const v8bf*)pl, c1 = *(const v8bf*)(pl + 16);
        Bh[nt] = __builtin_shufflevector(b0, b1, 0,1,2,3,4,5,6,7,8,9,10,11,12,13,14,15);
        Bl[nt] = __builtin_shufflevector(c0, c1, 0,1,2,3,4,5,6,7,8,9,10,11,12,13,14,15);
      }
      #pragma unroll
      for (int mtl = 0; mtl < 4; ++mtl) {
        const int m0 = rh * 64 + mtl * 16;
        asm volatile("" ::: "memory");     // keep A/gate loads local to tile
        // gates for this tile's 8 accumulator rows (broadcast reads)
        const float4 G0 = *(const float4*)&gatesT[e * TOKS + m0 + 8 * h];
        const float4 G1 = *(const float4*)&gatesT[e * TOKS + m0 + 8 * h + 4];
        // A fragments: row m0+l, K = kbc*32 + {8h..8h+7, 8h+16..8h+23}
        const int aO = (m0 + l) * HSTR + kbc * 32 + 8 * h;
        v8bf a0 = *(const v8bf*)&xhi[aO], a1 = *(const v8bf*)&xhi[aO + 16];
        v8bf e0 = *(const v8bf*)&xlo[aO], e1 = *(const v8bf*)&xlo[aO + 16];
        v16bf Ah = __builtin_shufflevector(a0, a1, 0,1,2,3,4,5,6,7,8,9,10,11,12,13,14,15);
        v16bf Al = __builtin_shufflevector(e0, e1, 0,1,2,3,4,5,6,7,8,9,10,11,12,13,14,15);
        #pragma unroll
        for (int nt = 0; nt < 2; ++nt) {
          v8f t = {};
          t = __builtin_amdgcn_wmma_f32_16x16x32_bf16(false, Ah, false, Bh[nt],
                                                      (short)0, t, false, false);
          t = __builtin_amdgcn_wmma_f32_16x16x32_bf16(false, Ah, false, Bl[nt],
                                                      (short)0, t, false, false);
          t = __builtin_amdgcn_wmma_f32_16x16x32_bf16(false, Al, false, Bh[nt],
                                                      (short)0, t, false, false);
          acc[mtl][nt][0] = fmaf(G0.x, t[0], acc[mtl][nt][0]);
          acc[mtl][nt][1] = fmaf(G0.y, t[1], acc[mtl][nt][1]);
          acc[mtl][nt][2] = fmaf(G0.z, t[2], acc[mtl][nt][2]);
          acc[mtl][nt][3] = fmaf(G0.w, t[3], acc[mtl][nt][3]);
          acc[mtl][nt][4] = fmaf(G1.x, t[4], acc[mtl][nt][4]);
          acc[mtl][nt][5] = fmaf(G1.y, t[5], acc[mtl][nt][5]);
          acc[mtl][nt][6] = fmaf(G1.z, t[6], acc[mtl][nt][6]);
          acc[mtl][nt][7] = fmaf(G1.w, t[7], acc[mtl][nt][7]);
        }
      }
    }
  }

  // ---------------- writeback (coalesced per 16-lane half) -------------------
  float* op = out + token0 * CDIM;
  #pragma unroll
  for (int mtl = 0; mtl < 4; ++mtl) {
    #pragma unroll
    for (int i = 0; i < 8; ++i) {
      const int row = rh * 64 + mtl * 16 + i + 8 * h;
      #pragma unroll
      for (int nt = 0; nt < 2; ++nt)
        op[(long)row * CDIM + n0 + nt * 16 + l] = acc[mtl][nt][i];
    }
  }
}

extern "C" void kernel_launch(void* const* d_in, const int* in_sizes, int n_in,
                              void* d_out, int out_size, void* d_ws, size_t ws_size,
                              hipStream_t stream) {
  (void)in_sizes; (void)n_in; (void)ws_size; (void)out_size;
  const float* x  = (const float*)d_in[0];
  const float* rw = (const float*)d_in[1];
  const float* rb = (const float*)d_in[2];
  const float* ew = (const float*)d_in[3];
  const float* eb = (const float*)d_in[4];
  float* out = (float*)d_out;

  __bf16* whi = (__bf16*)d_ws;                       // 256 KB
  __bf16* wlo = whi + (size_t)NEXP * DDIM * CDIM;    // 256 KB

  convert_w_kernel<<<(NEXP * DDIM * CDIM) / 256, 256, 0, stream>>>(ew, whi, wlo);

  const int tokens = 8 * 8192;             // B * N
  moe_fused_kernel<<<tokens / TOKS, 256, 0, stream>>>(x, rw, rb, eb, whi, wlo, out);
}
